// GemNetOutput_27041114095750
// MI455X (gfx1250) — compile-verified, matrix operand order checked by reading, not run on hardware
//
#include <hip/hip_runtime.h>
#include <hip/hip_bf16.h>
#include <math.h>

// ---------------------------------------------------------------------------
// Problem constants (match reference)
// ---------------------------------------------------------------------------
#define NN  1000000   // nodes
#define BB  16384     // graphs
#define HH  512       // hidden
#define FD  16        // film dim
#define H2  256       // hidden/2

typedef __attribute__((ext_vector_type(16))) __bf16        v16bf;
typedef __attribute__((ext_vector_type(8)))  float         v8f;
typedef __attribute__((ext_vector_type(8)))  unsigned int  v8u;

static __device__ __forceinline__ unsigned short f2bf(float f) {
    unsigned int u = __builtin_bit_cast(unsigned int, f);
    u += 0x7FFFu + ((u >> 16) & 1u);          // round-to-nearest-even
    return (unsigned short)(u >> 16);
}
static __device__ __forceinline__ float bf2f(unsigned short h) {
    return __builtin_bit_cast(float, (unsigned int)h << 16);
}
static __device__ __forceinline__ float silu(float x) {
    return x * (1.0f / (1.0f + __expf(-x)));
}

// ---------------------------------------------------------------------------
// 1) float -> bf16 weight conversion (grid-stride)
// ---------------------------------------------------------------------------
__global__ void cvt_bf16_kernel(const float* __restrict__ src,
                                unsigned short* __restrict__ dst, int n) {
    for (int i = blockIdx.x * blockDim.x + threadIdx.x; i < n;
         i += gridDim.x * blockDim.x)
        dst[i] = f2bf(src[i]);
}

// ---------------------------------------------------------------------------
// 2) Segment sum: x[N,512] summed by sorted batch[] into g[B,512].
//    HBM-roofline kernel (2 GB stream). 128 threads/block, each lane owns 4
//    consecutive columns (float4 loads -> 512B per wave per row). Register
//    accumulation while segment id unchanged; atomic f32 flush on boundary.
// ---------------------------------------------------------------------------
__global__ void segsum_kernel(const float* __restrict__ x,
                              const int*   __restrict__ batch,
                              float*       __restrict__ g) {
    const int col4 = threadIdx.x * 4;                  // 0..508
    const int rows = (NN + gridDim.x - 1) / gridDim.x;
    int lo = blockIdx.x * rows;
    int hi = lo + rows; if (hi > NN) hi = NN;
    if (lo >= hi) return;

    float4 acc = make_float4(0.f, 0.f, 0.f, 0.f);
    int cur = batch[lo];
    for (int r = lo; r < hi; ++r) {
        int bb = batch[r];
        if (bb != cur) {
            float* dst = &g[(size_t)cur * HH + col4];
            atomicAdd(dst + 0, acc.x);
            atomicAdd(dst + 1, acc.y);
            atomicAdd(dst + 2, acc.z);
            atomicAdd(dst + 3, acc.w);
            acc = make_float4(0.f, 0.f, 0.f, 0.f);
            cur = bb;
        }
        float4 v = *(const float4*)(x + (size_t)r * HH + col4);
        acc.x += v.x; acc.y += v.y; acc.z += v.z; acc.w += v.w;
    }
    float* dst = &g[(size_t)cur * HH + col4];
    atomicAdd(dst + 0, acc.x);
    atomicAdd(dst + 1, acc.y);
    atomicAdd(dst + 2, acc.z);
    atomicAdd(dst + 3, acc.w);
}

// ---------------------------------------------------------------------------
// 3) FiLM: gmod[b,h] = (emb[b,:]·gamma_w[h,:] + gamma_b[h]) * g[b,h]
//                    + (emb[b,:]·beta_w[h,:]  + beta_b[h])      -> bf16
// ---------------------------------------------------------------------------
__global__ void film_kernel(const float* __restrict__ g,
                            const float* __restrict__ emb,
                            const float* __restrict__ gw,
                            const float* __restrict__ gb,
                            const float* __restrict__ bw,
                            const float* __restrict__ bb,
                            unsigned short* __restrict__ gmod) {
    int idx = blockIdx.x * blockDim.x + threadIdx.x;
    if (idx >= BB * HH) return;
    int b = idx >> 9;          // /512
    int h = idx & (HH - 1);

    const float4* de = (const float4*)(emb + (size_t)b * FD);
    const float4* gr = (const float4*)(gw + (size_t)h * FD);
    const float4* br = (const float4*)(bw + (size_t)h * FD);

    float ga = gb[h], be = bb[h];
#pragma unroll
    for (int i = 0; i < 4; ++i) {
        float4 d = de[i], gv = gr[i], bv = br[i];
        ga += d.x * gv.x + d.y * gv.y + d.z * gv.z + d.w * gv.w;
        be += d.x * bv.x + d.y * bv.y + d.z * bv.z + d.w * bv.w;
    }
    gmod[idx] = f2bf(ga * g[idx] + be);
}

// ---------------------------------------------------------------------------
// 4) bf16 WMMA GEMM + bias + SiLU:  out[M,N] = silu(A[M,K] @ W[N,K]^T + bias)
//    One wave32 per 16x64 output supertile: one A fragment per K-step reused
//    across 4 v_wmma_f32_16x16x32_bf16 (4 f32 accumulators, 32 VGPRs).
//
//    16-bit A-fragment layout (ISA 7.12.2): lane L holds row M=L&15;
//    VGPR v holds K pair { kb + (v<4 ? 2v : 16+2(v-4)), +1 },
//    kb = (L>=16) ? 8 : 0 (two contiguous 16B runs -> b128-pair loads).
//    B mirrors A with N in place of M.
//    C/D: n = lane&15, m = (lane>>4)*8 + r for VGPR r.
// ---------------------------------------------------------------------------
__global__ void gemm_wmma_silu_kernel(const unsigned short* __restrict__ A,
                                      const unsigned short* __restrict__ W,
                                      const float*          __restrict__ bias,
                                      unsigned short*       __restrict__ out,
                                      int M, int N, int K) {
    const int wavesPerBlk = blockDim.x >> 5;
    const int wave = blockIdx.x * wavesPerBlk + (threadIdx.x >> 5);
    const int lane = threadIdx.x & 31;
    const int tilesN4 = N >> 6;                  // 64-wide supertiles
    const int tm = wave / tilesN4;
    const int tn = wave - tm * tilesN4;
    if (tm * 16 >= M) return;                    // wave-uniform

    const int l15 = lane & 15;
    const int kb  = (lane >> 4) << 3;            // 0 or 8
    const unsigned short* arow = A + (size_t)(tm * 16 + l15) * K;
    const unsigned short* wrow0 = W + (size_t)(tn * 64 + 0 * 16 + l15) * K;
    const unsigned short* wrow1 = W + (size_t)(tn * 64 + 1 * 16 + l15) * K;
    const unsigned short* wrow2 = W + (size_t)(tn * 64 + 2 * 16 + l15) * K;
    const unsigned short* wrow3 = W + (size_t)(tn * 64 + 3 * 16 + l15) * K;

    v8f c0 = {}, c1 = {}, c2 = {}, c3 = {};
    for (int k0 = 0; k0 < K; k0 += 32) {
        v8u au, bu0, bu1, bu2, bu3;
#pragma unroll
        for (int v = 0; v < 8; ++v) {
            const int k = k0 + kb + ((v < 4) ? (2 * v) : (16 + 2 * (v - 4)));
            au[v]  = *(const unsigned int*)(arow  + k);
            bu0[v] = *(const unsigned int*)(wrow0 + k);
            bu1[v] = *(const unsigned int*)(wrow1 + k);
            bu2[v] = *(const unsigned int*)(wrow2 + k);
            bu3[v] = *(const unsigned int*)(wrow3 + k);
        }
        v16bf a = __builtin_bit_cast(v16bf, au);
        c0 = __builtin_amdgcn_wmma_f32_16x16x32_bf16(
                false, a, false, __builtin_bit_cast(v16bf, bu0), (short)0, c0, false, false);
        c1 = __builtin_amdgcn_wmma_f32_16x16x32_bf16(
                false, a, false, __builtin_bit_cast(v16bf, bu1), (short)0, c1, false, false);
        c2 = __builtin_amdgcn_wmma_f32_16x16x32_bf16(
                false, a, false, __builtin_bit_cast(v16bf, bu2), (short)0, c2, false, false);
        c3 = __builtin_amdgcn_wmma_f32_16x16x32_bf16(
                false, a, false, __builtin_bit_cast(v16bf, bu3), (short)0, c3, false, false);
    }

    const int mbase = tm * 16 + ((lane >> 4) << 3);
    v8f cc[4] = {c0, c1, c2, c3};
#pragma unroll
    for (int j = 0; j < 4; ++j) {
        const int ncol = tn * 64 + j * 16 + l15;
        const float bn = bias[ncol];
#pragma unroll
        for (int r = 0; r < 8; ++r) {
            float v = silu(cc[j][r] + bn);
            out[(size_t)(mbase + r) * N + ncol] = f2bf(v);
        }
    }
}

// ---------------------------------------------------------------------------
// 5) Head: out[b] = h2[b,:] · w3 + b3 ; one wave per graph, shfl reduction.
// ---------------------------------------------------------------------------
__global__ void head_kernel(const unsigned short* __restrict__ h2,
                            const float* __restrict__ w3,
                            const float* __restrict__ b3,
                            float* __restrict__ out) {
    const int wave = blockIdx.x * (blockDim.x >> 5) + (threadIdx.x >> 5);
    const int lane = threadIdx.x & 31;
    if (wave >= BB) return;

    const unsigned short* row = h2 + (size_t)wave * H2;
    float acc = 0.0f;
#pragma unroll
    for (int i = 0; i < 8; ++i) {
        int k = lane * 8 + i;
        acc += bf2f(row[k]) * w3[k];
    }
#pragma unroll
    for (int off = 16; off > 0; off >>= 1)
        acc += __shfl_xor(acc, off, 32);
    if (lane == 0) out[wave] = acc + b3[0];
}

// ---------------------------------------------------------------------------
// Launch
// ---------------------------------------------------------------------------
extern "C" void kernel_launch(void* const* d_in, const int* in_sizes, int n_in,
                              void* d_out, int out_size, void* d_ws, size_t ws_size,
                              hipStream_t stream) {
    const float* x     = (const float*)d_in[0];
    const int*   batch = (const int*)  d_in[1];
    const float* emb   = (const float*)d_in[2];
    const float* gw    = (const float*)d_in[3];
    const float* gb    = (const float*)d_in[4];
    const float* bw    = (const float*)d_in[5];
    const float* bbv   = (const float*)d_in[6];
    const float* w1    = (const float*)d_in[7];
    const float* b1    = (const float*)d_in[8];
    const float* w2    = (const float*)d_in[9];
    const float* b2    = (const float*)d_in[10];
    const float* w3    = (const float*)d_in[11];
    const float* b3    = (const float*)d_in[12];
    float* out = (float*)d_out;

    char* ws = (char*)d_ws;
    size_t off = 0;
    auto alloc = [&](size_t bytes) {
        char* p = ws + off;
        off += (bytes + 255) & ~(size_t)255;
        return p;
    };
    float*          g    = (float*)         alloc((size_t)BB * HH * 4);  // 32 MB
    unsigned short* gmod = (unsigned short*)alloc((size_t)BB * HH * 2);  // 16 MB
    unsigned short* w1b  = (unsigned short*)alloc((size_t)HH * HH * 2);  // 512 KB
    unsigned short* w2b  = (unsigned short*)alloc((size_t)H2 * HH * 2);  // 256 KB
    unsigned short* h1   = (unsigned short*)alloc((size_t)BB * HH * 2);  // 16 MB
    unsigned short* h2   = (unsigned short*)alloc((size_t)BB * H2 * 2);  //  8 MB
    (void)ws_size; (void)in_sizes; (void)n_in; (void)out_size;

    // zero segment-sum accumulator
    hipMemsetAsync(g, 0, (size_t)BB * HH * sizeof(float), stream);

    // weight conversion to bf16
    cvt_bf16_kernel<<<512, 256, 0, stream>>>(w1, w1b, HH * HH);
    cvt_bf16_kernel<<<256, 256, 0, stream>>>(w2, w2b, H2 * HH);

    // segment sum (HBM-bound: 2 GB stream of x), float4 per lane
    segsum_kernel<<<4096, 128, 0, stream>>>(x, batch, g);

    // FiLM modulation -> bf16 activations
    film_kernel<<<(BB * HH) / 256, 256, 0, stream>>>(g, emb, gw, gb, bw, bbv, gmod);

    // MLP head via bf16 WMMA, 16x64 output per wave, 4 waves per block
    {
        int waves = (BB / 16) * (HH / 64);            // 8192
        gemm_wmma_silu_kernel<<<waves / 4, 128, 0, stream>>>(gmod, w1b, b1, h1, BB, HH, HH);
    }
    {
        int waves = (BB / 16) * (H2 / 64);            // 4096
        gemm_wmma_silu_kernel<<<waves / 4, 128, 0, stream>>>(h1, w2b, b2, h2, BB, H2, HH);
    }

    // final projection to scalar per graph
    head_kernel<<<BB / 4, 128, 0, stream>>>(h2, w3, b3, out);
}
